// FoldingNet_32633161515576
// MI455X (gfx1250) — matrix operand order repaired and checked by reading, not compiled, not run on hardware
//
#include <hip/hip_runtime.h>
#include <hip/hip_bf16.h>

// ---------------------------------------------------------------------------
// FoldingNet encoder for MI455X (gfx1250, wave32, WMMA).
//  - All channel-mixing GEMMs on v_wmma_f32_16x16x32_f16, KPAD templated so
//    the K loop fully unrolls and fragments load as b128 vectors.
//  - Weights pre-converted fp32->f16 into K-padded workspace: B fragment is
//    16 contiguous halfs per lane -> 2x global_load_b128, zero branches.
//  - Global max over N fused into the 128->1024 GEMM epilogue (atomic max on
//    order-preserving uint encoding -> deterministic).
// ---------------------------------------------------------------------------

#define BN   32
#define NPTS 2048
#define KNN  16
#define ROWS (BN * NPTS)   // 65536 points (batch folded into GEMM M)

typedef __attribute__((ext_vector_type(16))) _Float16 v16h;
typedef __attribute__((ext_vector_type(8)))  _Float16 v8h;
typedef __attribute__((ext_vector_type(8)))  float    v8f;

// ---- order-preserving float<->uint encoding for atomic max -----------------
__device__ __forceinline__ unsigned enc_f32(float f) {
    unsigned u = __float_as_uint(f);
    return (u & 0x80000000u) ? ~u : (u | 0x80000000u);
}
__device__ __forceinline__ float dec_f32(unsigned u) {
    return (u & 0x80000000u) ? __uint_as_float(u & 0x7FFFFFFFu)
                             : __uint_as_float(~u);
}

// ---------------------------------------------------------------------------
// 1) KNN: batch points staged in 24KB LDS; register top-16 per query.
// ---------------------------------------------------------------------------
__global__ __launch_bounds__(128)
void knn_kernel(const float* __restrict__ x, int* __restrict__ idx) {
    __shared__ float sx[NPTS], sy[NPTS], sz[NPTS];
    int b = blockIdx.y;
    const float* xb = x + (size_t)b * NPTS * 3;
    for (int i = threadIdx.x; i < NPTS; i += blockDim.x) {
        sx[i] = xb[i * 3 + 0];
        sy[i] = xb[i * 3 + 1];
        sz[i] = xb[i * 3 + 2];
    }
    __syncthreads();

    int q = blockIdx.x * blockDim.x + threadIdx.x;
    float px = sx[q], py = sy[q], pz = sz[q];

    float bd[KNN]; int bi[KNN];
#pragma unroll
    for (int i = 0; i < KNN; ++i) { bd[i] = -3.4e38f; bi[i] = 0; }

    for (int j = 0; j < NPTS; ++j) {
        float dx = px - sx[j], dy = py - sy[j], dz = pz - sz[j];
        float d = -(dx * dx + dy * dy + dz * dz);
        if (d > bd[KNN - 1]) {
            float cd = d; int ci = j;
#pragma unroll
            for (int t = 0; t < KNN; ++t) {
                if (cd > bd[t]) {
                    float td = bd[t]; int ti = bi[t];
                    bd[t] = cd; bi[t] = ci;
                    cd = td; ci = ti;
                }
            }
        }
    }
    int* op = idx + ((size_t)b * NPTS + q) * KNN;
#pragma unroll
    for (int i = 0; i < KNN; ++i) op[i] = bi[i];
}

// ---------------------------------------------------------------------------
// 2) 12-channel features [pts(3) | p0 outer p1 (9)], stride 32 (K padding).
// ---------------------------------------------------------------------------
__global__ __launch_bounds__(256)
void feat_kernel(const float* __restrict__ x, const int* __restrict__ idx,
                 _Float16* __restrict__ feat) {
    int row = blockIdx.x * blockDim.x + threadIdx.x;
    if (row >= ROWS) return;
    int b = row / NPTS;
    const float* xr = x + (size_t)row * 3;
    float p[3] = {xr[0], xr[1], xr[2]};
    int i0 = idx[(size_t)row * KNN + 0];
    int i1 = idx[(size_t)row * KNN + 1];
    const float* x0 = x + ((size_t)b * NPTS + i0) * 3;
    const float* x1 = x + ((size_t)b * NPTS + i1) * 3;
    float p0[3] = {x0[0], x0[1], x0[2]};
    float p1[3] = {x1[0], x1[1], x1[2]};

    _Float16* f = feat + (size_t)row * 32;
    f[0] = (_Float16)p[0]; f[1] = (_Float16)p[1]; f[2] = (_Float16)p[2];
#pragma unroll
    for (int i = 0; i < 3; ++i)
#pragma unroll
        for (int j = 0; j < 3; ++j)
            f[3 + 3 * i + j] = (_Float16)(p0[i] * p1[j]);
#pragma unroll
    for (int c = 12; c < 32; ++c) f[c] = (_Float16)0.f;
}

// ---------------------------------------------------------------------------
// 3) Weight conversion: fp32 [Cout][Cin] -> f16 [Cout][Kpad] (zero padded).
// ---------------------------------------------------------------------------
__global__ __launch_bounds__(256)
void wconv_kernel(const float* __restrict__ W, _Float16* __restrict__ W16,
                  int Cin, int Kpad, int total) {
    int i = blockIdx.x * blockDim.x + threadIdx.x;
    if (i >= total) return;
    int o = i / Kpad, k = i - o * Kpad;
    W16[i] = (k < Cin) ? (_Float16)W[o * Cin + k] : (_Float16)0.f;
}

// ---------------------------------------------------------------------------
// WMMA fragment helpers (wave32, k = 16*half + s in both A and B, all loads 16B-aligned vectors).
// A 16x32 f16: lane m=lane&15, two contiguous 8-half runs per K-step.
// B 32x16 f16: lane's 16 halfs are K-contiguous -> single v16h load.
// ---------------------------------------------------------------------------
__device__ __forceinline__ v16h a_frag(const _Float16* As, int stride,
                                       int m, int half, int k0) {
    v8h lo = *(const v8h*)(As + m * stride + k0 + 16 * half);
    v8h hi = *(const v8h*)(As + m * stride + k0 + 16 * half + 8);
    return __builtin_shufflevector(lo, hi, 0, 1, 2, 3, 4, 5, 6, 7,
                                           8, 9, 10, 11, 12, 13, 14, 15);
}

// ---------------------------------------------------------------------------
// 4) WMMA GEMM: Out[M=pts][N=Cout] = A[M][KPAD] * W16^T + bias (opt ReLU).
//    Block = 4 waves sharing one LDS A tile; wave owns a 16x16 C tile.
// ---------------------------------------------------------------------------
template <int KPAD, int RELU, bool OUTF32>
__global__ __launch_bounds__(128)
void gemm_kernel(const _Float16* __restrict__ A,
                 const _Float16* __restrict__ W16,
                 const float* __restrict__ bias,
                 int Cout, void* __restrict__ outp) {
    __shared__ _Float16 As[16 * KPAD];
    int rowbase = blockIdx.x * 16;
    {
        const v8h* src = (const v8h*)(A + (size_t)rowbase * KPAD);
        v8h* dst = (v8h*)As;
        constexpr int CH = 16 * KPAD / 8;
#pragma unroll
        for (int i = threadIdx.x; i < CH; i += 128) dst[i] = src[i];
    }
    __syncthreads();

    int lane = threadIdx.x & 31;
    int half = lane >> 4;
    int m = lane & 15;
    int cg = (blockIdx.y * 4 + (threadIdx.x >> 5)) * 16 + m;
    const _Float16* wrow = W16 + (size_t)cg * KPAD;
    __builtin_prefetch(wrow, 0, 3);

    v8f acc = {};
#pragma unroll
    for (int k0 = 0; k0 < KPAD; k0 += 32) {
        v16h a = a_frag(As, KPAD, m, half, k0);
        v16h bf = *(const v16h*)(wrow + k0 + 16 * half);
        acc = __builtin_amdgcn_wmma_f32_16x16x32_f16(
            false, a, false, bf, (short)0, acc, false, false);
    }

    float bv = bias[cg];
#pragma unroll
    for (int r = 0; r < 8; ++r) {
        float v = acc[r] + bv;
        if (RELU) v = fmaxf(v, 0.f);
        size_t o = (size_t)(rowbase + r + 8 * half) * Cout + cg;
        if (OUTF32) ((float*)outp)[o] = v;
        else        ((_Float16*)outp)[o] = (_Float16)v;
    }
}

// ---------------------------------------------------------------------------
// 5) 128 -> 1024 conv with global max-over-N fused (never materializes the
//    256MB (B,1024,N) tensor). Atomic max on ordered uints: deterministic.
// ---------------------------------------------------------------------------
__global__ __launch_bounds__(128)
void gemm_max_kernel(const _Float16* __restrict__ A,
                     const _Float16* __restrict__ W16,
                     const float* __restrict__ bias,
                     unsigned* __restrict__ g_u) {
    constexpr int KPAD = 128;
    __shared__ _Float16 As[16 * KPAD];
    int rowbase = blockIdx.x * 16;
    int b = rowbase / NPTS;
    {
        const v8h* src = (const v8h*)(A + (size_t)rowbase * KPAD);
        v8h* dst = (v8h*)As;
#pragma unroll
        for (int i = threadIdx.x; i < 16 * KPAD / 8; i += 128) dst[i] = src[i];
    }
    __syncthreads();

    int lane = threadIdx.x & 31;
    int half = lane >> 4;
    int m = lane & 15;
    int cg = (blockIdx.y * 4 + (threadIdx.x >> 5)) * 16 + m;
    const _Float16* wrow = W16 + (size_t)cg * KPAD;
    __builtin_prefetch(wrow, 0, 3);

    v8f acc = {};
#pragma unroll
    for (int k0 = 0; k0 < KPAD; k0 += 32) {
        v16h a = a_frag(As, KPAD, m, half, k0);
        v16h bf = *(const v16h*)(wrow + k0 + 16 * half);
        acc = __builtin_amdgcn_wmma_f32_16x16x32_f16(
            false, a, false, bf, (short)0, acc, false, false);
    }

    float bv = bias[cg];
    float mx = -3.4e38f;
#pragma unroll
    for (int r = 0; r < 8; ++r) mx = fmaxf(mx, acc[r] + bv);
    atomicMax(&g_u[b * 1024 + cg], enc_f32(mx));
}

// ---------------------------------------------------------------------------
// 6) Gather 16 neighbors + per-channel max (memory-bound, C templated).
// ---------------------------------------------------------------------------
template <int C>
__global__ __launch_bounds__(256)
void gathermax_kernel(const _Float16* __restrict__ A,
                      const int* __restrict__ idx,
                      _Float16* __restrict__ Out) {
    long t = (long)blockIdx.x * blockDim.x + threadIdx.x;  // over ROWS*C
    int c = (int)(t % C);
    long row = t / C;
    if (row >= ROWS) return;
    long b = row / NPTS;
    const int* ip = idx + row * KNN;
    float m = -3.4e38f;
#pragma unroll
    for (int k = 0; k < KNN; ++k) {
        long j = ip[k];
        m = fmaxf(m, (float)A[(b * NPTS + j) * C + c]);
    }
    Out[row * C + c] = (_Float16)m;
}

__global__ __launch_bounds__(256)
void initg_kernel(unsigned* __restrict__ g_u) {
    int i = blockIdx.x * blockDim.x + threadIdx.x;
    if (i < BN * 1024) g_u[i] = 0u;   // below every real encoded float
}

__global__ __launch_bounds__(256)
void decode_kernel(const unsigned* __restrict__ g_u, _Float16* __restrict__ g) {
    int i = blockIdx.x * blockDim.x + threadIdx.x;
    if (i < BN * 1024) g[i] = (_Float16)dec_f32(g_u[i]);
}

// ---------------------------------------------------------------------------
// Host side
// ---------------------------------------------------------------------------
extern "C" void kernel_launch(void* const* d_in, const int* in_sizes, int n_in,
                              void* d_out, int out_size, void* d_ws, size_t ws_size,
                              hipStream_t stream) {
    (void)in_sizes; (void)n_in; (void)out_size; (void)ws_size;

    const float* x     = (const float*)d_in[0];
    const float* w_m1a = (const float*)d_in[1];
    const float* b_m1a = (const float*)d_in[2];
    const float* w_m1b = (const float*)d_in[3];
    const float* b_m1b = (const float*)d_in[4];
    const float* w_m1c = (const float*)d_in[5];
    const float* b_m1c = (const float*)d_in[6];
    const float* w_l1  = (const float*)d_in[7];
    const float* b_l1  = (const float*)d_in[8];
    const float* w_c1  = (const float*)d_in[9];
    const float* b_c1  = (const float*)d_in[10];
    const float* w_l2  = (const float*)d_in[11];
    const float* b_l2  = (const float*)d_in[12];
    const float* w_c2  = (const float*)d_in[13];
    const float* b_c2  = (const float*)d_in[14];
    const float* w_m2a = (const float*)d_in[15];
    const float* b_m2a = (const float*)d_in[16];
    const float* w_m2b = (const float*)d_in[17];
    const float* b_m2b = (const float*)d_in[18];

    char* ws = (char*)d_ws;
    int*      idx  = (int*)(ws);                        //  4 MB
    _Float16* bufA = (_Float16*)(ws + (4u  << 20));     // 16 MB
    _Float16* bufB = (_Float16*)(ws + (20u << 20));     // 16 MB
    unsigned* g_u  = (unsigned*)(ws + (36u << 20));     // 128 KB
    _Float16* g1   = (_Float16*)(ws + (37u << 20));     //  64 KB
    _Float16* g2   = (_Float16*)(ws + (38u << 20));     //  64 KB
    // f16 K-padded weights
    char* wb = ws + (39u << 20);
    _Float16* W_m1a = (_Float16*)(wb);                       // 64*32
    _Float16* W_m1b = (_Float16*)(wb + (64u  << 10));        // 64*64
    _Float16* W_m1c = (_Float16*)(wb + (128u << 10));
    _Float16* W_l1  = (_Float16*)(wb + (192u << 10));
    _Float16* W_c1  = (_Float16*)(wb + (256u << 10));        // 128*64
    _Float16* W_l2  = (_Float16*)(wb + (320u << 10));        // 128*128
    _Float16* W_c2  = (_Float16*)(wb + (384u << 10));        // 1024*128
    _Float16* W_m2a = (_Float16*)(wb + (640u << 10));        // 1024*1024
    _Float16* W_m2b = (_Float16*)(wb + (640u << 10) + (2u << 21));

    // ---- weight conversion (fp32 -> f16, K-padded) ----
    wconv_kernel<<<8,    256, 0, stream>>>(w_m1a, W_m1a, 12,   32,   64 * 32);
    wconv_kernel<<<16,   256, 0, stream>>>(w_m1b, W_m1b, 64,   64,   64 * 64);
    wconv_kernel<<<16,   256, 0, stream>>>(w_m1c, W_m1c, 64,   64,   64 * 64);
    wconv_kernel<<<16,   256, 0, stream>>>(w_l1,  W_l1,  64,   64,   64 * 64);
    wconv_kernel<<<32,   256, 0, stream>>>(w_c1,  W_c1,  64,   64,   128 * 64);
    wconv_kernel<<<64,   256, 0, stream>>>(w_l2,  W_l2,  128,  128,  128 * 128);
    wconv_kernel<<<512,  256, 0, stream>>>(w_c2,  W_c2,  128,  128,  1024 * 128);
    wconv_kernel<<<4096, 256, 0, stream>>>(w_m2a, W_m2a, 1024, 1024, 1024 * 1024);
    wconv_kernel<<<4096, 256, 0, stream>>>(w_m2b, W_m2b, 1024, 1024, 1024 * 1024);

    // ---- KNN + features ----
    knn_kernel <<<dim3(NPTS / 128, BN), 128, 0, stream>>>(x, idx);
    feat_kernel<<<ROWS / 256, 256, 0, stream>>>(x, idx, bufA);

    // ---- MLP1: 12->64->64->64 (ReLU) ----
    gemm_kernel<32, 1, false><<<dim3(ROWS / 16, 1), 128, 0, stream>>>(bufA, W_m1a, b_m1a, 64, bufB);
    gemm_kernel<64, 1, false><<<dim3(ROWS / 16, 1), 128, 0, stream>>>(bufB, W_m1b, b_m1b, 64, bufA);
    gemm_kernel<64, 1, false><<<dim3(ROWS / 16, 1), 128, 0, stream>>>(bufA, W_m1c, b_m1c, 64, bufB);

    // ---- local max pool + linear 64->64 ----
    gathermax_kernel<64><<<(ROWS * 64) / 256, 256, 0, stream>>>(bufB, idx, bufA);
    gemm_kernel<64, 0, false><<<dim3(ROWS / 16, 1), 128, 0, stream>>>(bufA, W_l1, b_l1, 64, bufB);

    // ---- conv 64->128 (ReLU), local max pool, linear 128->128 ----
    gemm_kernel<64, 1, false><<<dim3(ROWS / 16, 2), 128, 0, stream>>>(bufB, W_c1, b_c1, 128, bufA);
    gathermax_kernel<128><<<(ROWS * 128) / 256, 256, 0, stream>>>(bufA, idx, bufB);
    gemm_kernel<128, 0, false><<<dim3(ROWS / 16, 2), 128, 0, stream>>>(bufB, W_l2, b_l2, 128, bufA);

    // ---- conv 128->1024 fused with global max over N ----
    initg_kernel<<<(BN * 1024) / 256, 256, 0, stream>>>(g_u);
    gemm_max_kernel<<<dim3(ROWS / 16, 16), 128, 0, stream>>>(bufA, W_c2, b_c2, g_u);
    decode_kernel<<<(BN * 1024) / 256, 256, 0, stream>>>(g_u, g1);

    // ---- head: 1024->1024 (ReLU) -> 1024->1024, fp32 out (B,1,1024) ----
    gemm_kernel<1024, 1, false><<<dim3(2, 16), 128, 0, stream>>>(g1, W_m2a, b_m2a, 1024, g2);
    gemm_kernel<1024, 0, true ><<<dim3(2, 16), 128, 0, stream>>>(g2, W_m2b, b_m2b, 1024, (void*)d_out);
}
